// SpikingSSMLayer_33071248179947
// MI455X (gfx1250) — matrix operand order repaired and checked by reading, not compiled, not run on hardware
//
#include <hip/hip_runtime.h>
#include <hip/hip_bf16.h>

// ---------------------------------------------------------------------------
// SpikingSSMLayer for MI455X (gfx1250, wave32, WMMA + TDM)
//
// Shapes (fixed by reference): B=8, T=4, L=512, D=1024, N=64, R=64
// Pipeline per timestep: LIF_in -> dt GEMM(WMMA)+softplus -> dA/dtm(exp) ->
//                        u GEMM(WMMA) -> h scan (tiny sequential) ->
//                        y GEMM(WMMA)+Dv*xs -> LIF_out
//
// GEMMs: weight tiles staged into LDS by the Tensor Data Mover
// (tensor_load_to_lds, TENSORcnt-tracked, double-buffered); A fragments
// software-pipelined in registers; B fragments batch-loaded from LDS so one
// dscnt wait covers a back-to-back group of four v_wmma ops.
// ---------------------------------------------------------------------------

#define B_   8
#define T_   4
#define L_   512
#define D_   1024
#define N_   64
#define R_   64
#define ROWS (B_ * L_)   /* 4096 */

typedef __attribute__((ext_vector_type(16))) _Float16 v16h;
typedef __attribute__((ext_vector_type(8)))  _Float16 v8h;
typedef __attribute__((ext_vector_type(8)))  float    v8f;
typedef __attribute__((ext_vector_type(4)))  unsigned int v4u;
typedef __attribute__((ext_vector_type(8)))  int      v8i;
typedef __attribute__((ext_vector_type(4)))  int      v4i;

// ---------------------------------------------------------------------------
// TDM: DMA a (64 rows x 32 cols) f16 tile (row stride = K elements) from
// global memory into LDS at lds_byte_addr. D# packed per CDNA5 ISA ch.8.
// ---------------------------------------------------------------------------
__device__ __forceinline__ void tdm_load_tile(const _Float16* gsrc,
                                              unsigned lds_byte_addr, int K)
{
    unsigned long long ga = (unsigned long long)(uintptr_t)gsrc;
    v4u g0;
    g0[0] = 1u;                                   // count=1, is_restore=0
    g0[1] = lds_byte_addr;                        // lds_addr [63:32]
    g0[2] = (unsigned)(ga & 0xFFFFFFFFu);         // global_addr lo
    g0[3] = (unsigned)((ga >> 32) & 0x01FFFFFFu)  // global_addr hi (bits 56:32)
          | (2u << 30);                           // type=2 ("image")
    v8i g1;
    g1[0] = (int)(1u << 16);                      // wg_mask=0, data_size=1 (2B)
    g1[1] = (int)(((unsigned)K & 0xFFFFu) << 16); // tensor_dim0 lo16 (<<16)
    g1[2] = (int)((((unsigned)K >> 16) & 0xFFFFu) // tensor_dim0 hi16
          | (64u << 16));                         // tensor_dim1 lo16 = 64 rows
    g1[3] = (int)(32u << 16);                     // tensor_dim1 hi=0 | tile_dim0=32
    g1[4] = 64;                                   // tile_dim1=64, tile_dim2=0
    g1[5] = K;                                    // tensor_dim0_stride lo32
    g1[6] = 0;                                    // stride hi | dim1_stride lo
    g1[7] = 0;
    v4i z4 = {0, 0, 0, 0};
#if __clang_major__ >= 23
    v8i z8 = {0, 0, 0, 0, 0, 0, 0, 0};
    __builtin_amdgcn_tensor_load_to_lds(g0, g1, z4, z4, z8, 0);
#else
    __builtin_amdgcn_tensor_load_to_lds(g0, g1, z4, z4, 0);
#endif
}

// ---------------------------------------------------------------------------
// A fragment (16x32 f16): lane lm holds row m; K = kh*16+[0..15]
// ---------------------------------------------------------------------------
__device__ __forceinline__ v16h load_a_frag(const _Float16* __restrict__ arow,
                                            int k0, int kh)
{
    v8h alo = *(const v8h*)(arow + k0 + kh * 16);
    v8h ahi = *(const v8h*)(arow + k0 + kh * 16 + 8);
    v16h a;
    #pragma unroll
    for (int i = 0; i < 8; ++i) { a[i] = alo[i]; a[i + 8] = ahi[i]; }
    return a;
}

// ---------------------------------------------------------------------------
// WMMA GEMM: out(MxN) = A(MxK) * W(NxK)^T, f16 in / f32 accumulate.
// Block = 256 threads = 8 waves -> 128 rows x 64 cols.
// EPI 0: softplus(acc + e0[col])      (dt path)
// EPI 1: plain store                  (u path)
// EPI 2: acc + e0[col]*e1[row*N+col]  (y path: Dv * xs)
// ---------------------------------------------------------------------------
template<int EPI>
__global__ void __launch_bounds__(256)
wmma_gemm_nt(const _Float16* __restrict__ A,
             const _Float16* __restrict__ W,
             float* __restrict__ out,
             const float* __restrict__ e0,
             const float* __restrict__ e1,
             int M, int N, int K)
{
    __shared__ _Float16 sW[2][64 * 32];           // 2 x 4KB double buffer

    const int wave = threadIdx.x >> 5;
    const int lane = threadIdx.x & 31;
    const int m0   = blockIdx.x * 128 + wave * 16;
    const int n0   = blockIdx.y * 64;
    const int lm   = lane & 15;   // row within A tile / col within B tile
    const int kh   = lane >> 4;   // K-half select

    const _Float16* __restrict__ arow = A + (size_t)(m0 + lm) * K;
    const _Float16* __restrict__ wtile = W + (size_t)n0 * K;

    const int nk = K >> 5;        // K-steps of 32 (>= 2 in all launches)
    v8f acc[4] = {};

    // ---- prologue: TDM tile 0 + first A fragment ----
    if (wave == 0) {
        tdm_load_tile(wtile, (unsigned)(uintptr_t)&sW[0][0], K);
        __builtin_amdgcn_s_wait_tensorcnt(0);
    }
    v16h afrag = load_a_frag(arow, 0, kh);
    __syncthreads();

    // ---- steady state: unconditional issue/prefetch/compute/sync ----
    int buf = 0;
    #pragma unroll 1
    for (int kc = 0; kc < nk - 1; ++kc) {
        if (wave == 0)
            tdm_load_tile(wtile + (kc + 1) * 32,
                          (unsigned)(uintptr_t)&sW[buf ^ 1][0], K);
        v16h anext = load_a_frag(arow, (kc + 1) * 32, kh);

        // batch all four B fragments (one dscnt wait covers the MMA group)
        v16h bf[4];
        #pragma unroll
        for (int t = 0; t < 4; ++t)
            bf[t] = *(const v16h*)&sW[buf][(t * 16 + lm) * 32 + kh * 16];

        #pragma unroll
        for (int t = 0; t < 4; ++t)
            acc[t] = __builtin_amdgcn_wmma_f32_16x16x32_f16(
                false, afrag, false, bf[t], (short)0, acc[t], false, false);

        afrag = anext;
        if (wave == 0) __builtin_amdgcn_s_wait_tensorcnt(0);
        __syncthreads();
        buf ^= 1;
    }

    // ---- peeled final K-step ----
    {
        v16h bf[4];
        #pragma unroll
        for (int t = 0; t < 4; ++t)
            bf[t] = *(const v16h*)&sW[buf][(t * 16 + lm) * 32 + kh * 16];
        #pragma unroll
        for (int t = 0; t < 4; ++t)
            acc[t] = __builtin_amdgcn_wmma_f32_16x16x32_f16(
                false, afrag, false, bf[t], (short)0, acc[t], false, false);
    }

    // C/D layout: VGPR i, lanes 0-15 -> M=i, lanes 16-31 -> M=i+8; N = lane%16
    const int rbase = m0 + kh * 8;
    #pragma unroll
    for (int t = 0; t < 4; ++t) {
        const int col = n0 + t * 16 + lm;
        #pragma unroll
        for (int i = 0; i < 8; ++i) {
            const int row = rbase + i;
            float v = acc[t][i];
            if (EPI == 0) {
                v += e0[col];
                v = fmaxf(v, 0.f) + log1pf(__expf(-fabsf(v)));  // softplus
            } else if (EPI == 2) {
                v = fmaf(e0[col], e1[(size_t)row * N + col], v);
            }
            out[(size_t)row * N + col] = v;
        }
    }
}

// ---------------------------------------------------------------------------
// LIF neurons (MultiThresholdLIF forward, NUM_THR = 3)
// ---------------------------------------------------------------------------
__device__ __forceinline__ float lif_core(float xi, float* vp, float* ap,
                                          const float* __restrict__ tau,
                                          const float* __restrict__ thr,
                                          const float* __restrict__ rst,
                                          int d)
{
    float vv = *vp, aa = *ap;
    float decay = __expf(-1.0f / fmaxf(tau[d], 0.1f));
    vv = vv * decay + xi - 0.1f * aa;
    float spikes = 0.f;
    #pragma unroll
    for (int i = 0; i < 3; ++i) {
        float s = (vv >= thr[d * 3 + i]) ? 1.f : 0.f;
        spikes += s * (float)(i + 1);
        vv = vv * (1.f - s) + rst[d * 3 + i] * s;
    }
    aa = aa * 0.95f + ((spikes > 0.f) ? 0.05f : 0.f);
    *vp = vv; *ap = aa;
    return spikes * (1.f / 3.f);
}

__global__ void __launch_bounds__(256)
lif_in_kernel(const float* __restrict__ x, int t,
              float* __restrict__ v, float* __restrict__ a,
              const float* __restrict__ tau, const float* __restrict__ thr,
              const float* __restrict__ rst,
              float* __restrict__ xs, _Float16* __restrict__ xs16)
{
    int idx = blockIdx.x * 256 + threadIdx.x;      // < ROWS*D
    int row = idx >> 10, d = idx & (D_ - 1);
    int b = row >> 9,  l = row & (L_ - 1);
    float xi = x[(((size_t)(b * T_ + t) * L_ + l) << 10) + d];
    float vv = v[idx], aa = a[idx];
    float o = lif_core(xi, &vv, &aa, tau, thr, rst, d);
    v[idx] = vv; a[idx] = aa;
    xs[idx]   = o;
    xs16[idx] = (_Float16)o;
}

__global__ void __launch_bounds__(256)
lif_out_kernel(const float* __restrict__ y, int t,
               float* __restrict__ v, float* __restrict__ a,
               const float* __restrict__ tau, const float* __restrict__ thr,
               const float* __restrict__ rst,
               float* __restrict__ out)
{
    int idx = blockIdx.x * 256 + threadIdx.x;
    int row = idx >> 10, d = idx & (D_ - 1);
    int b = row >> 9,  l = row & (L_ - 1);
    float vv = v[idx], aa = a[idx];
    float o = lif_core(y[idx], &vv, &aa, tau, thr, rst, d);
    v[idx] = vv; a[idx] = aa;
    out[(((size_t)(b * T_ + t) * L_ + l) << 10) + d] = o;
}

// ---------------------------------------------------------------------------
// dA[row,n] = mean_r exp(dt[row,r]*A[n]); dtm[row] = mean_r dt[row,r]
// Block handles 4 rows; thread (rlocal, n). 64 v_exp_f32 per thread.
// ---------------------------------------------------------------------------
__global__ void __launch_bounds__(256)
da_kernel(const float* __restrict__ dt, const float* __restrict__ A,
          float* __restrict__ dA, float* __restrict__ dtm)
{
    __shared__ float sdt[4][64];
    __shared__ float sA[64];
    int t = threadIdx.x;
    int rl = t >> 6, n = t & 63;
    int row = blockIdx.x * 4 + rl;
    if (t < 64) sA[t] = A[t];
    sdt[rl][n] = dt[row * 64 + n];
    __syncthreads();
    float an = sA[n];
    float s = 0.f, sm = 0.f;
    #pragma unroll 8
    for (int r = 0; r < 64; ++r) {
        float dv = sdt[rl][r];
        s  += __expf(dv * an);
        sm += dv;
    }
    dA[row * 64 + n] = s * (1.f / 64.f);
    if (n == 0) dtm[row] = sm * (1.f / 64.f);
}

// ---------------------------------------------------------------------------
// Sequential h recurrence over L. State is tiny (B=8 x N=64); one block per b.
// ---------------------------------------------------------------------------
__global__ void hscan_kernel(const float* __restrict__ dA,
                             const float* __restrict__ dtm,
                             const float* __restrict__ u,
                             _Float16* __restrict__ H16)
{
    int b = blockIdx.x, n = threadIdx.x;   // 64 threads
    float h = 0.f;
    for (int l = 0; l < L_; ++l) {
        int row = b * L_ + l;
        h = dA[row * 64 + n] * h + dtm[row] * u[row * 64 + n];
        H16[row * 64 + n] = (_Float16)h;
    }
}

// ---------------------------------------------------------------------------
// Utilities
// ---------------------------------------------------------------------------
__global__ void __launch_bounds__(256)
zero_kernel(float4* __restrict__ p)
{
    p[blockIdx.x * 256 + threadIdx.x] = make_float4(0.f, 0.f, 0.f, 0.f);
}

__global__ void __launch_bounds__(256)
f32_to_f16_kernel(const float* __restrict__ src, _Float16* __restrict__ dst)
{
    int i = blockIdx.x * 256 + threadIdx.x;
    dst[i] = (_Float16)src[i];
}

// ---------------------------------------------------------------------------
extern "C" void kernel_launch(void* const* d_in, const int* in_sizes, int n_in,
                              void* d_out, int out_size, void* d_ws, size_t ws_size,
                              hipStream_t stream)
{
    const float* x      = (const float*)d_in[0];   // (B,T,L,D)
    const float* A      = (const float*)d_in[1];   // (N,)
    const float* Bm     = (const float*)d_in[2];   // (N,D)
    const float* Cm     = (const float*)d_in[3];   // (D,N)
    const float* Dv     = (const float*)d_in[4];   // (D,)
    const float* Wdt    = (const float*)d_in[5];   // (R,D)
    const float* dtb    = (const float*)d_in[6];   // (R,)
    const float* tau_i  = (const float*)d_in[7];
    const float* thr_i  = (const float*)d_in[8];
    const float* rst_i  = (const float*)d_in[9];
    const float* tau_o  = (const float*)d_in[10];
    const float* thr_o  = (const float*)d_in[11];
    const float* rst_o  = (const float*)d_in[12];
    float* out = (float*)d_out;

    // ---- workspace carve-up (all 256B aligned; total ~108 MB) ----
    char* wp = (char*)d_ws;
    auto carve = [&](size_t bytes) -> char* {
        char* p = wp; wp += (bytes + 255) & ~(size_t)255; return p;
    };
    const size_t elems = (size_t)ROWS * D_;        // 4 Mi
    float*    v_i  = (float*)carve(elems * 4);     // LIF states (contiguous,
    float*    a_i  = (float*)carve(elems * 4);     //  zeroed together below)
    float*    v_o  = (float*)carve(elems * 4);
    float*    a_o  = (float*)carve(elems * 4);
    float*    xs   = (float*)carve(elems * 4);     // spike activations f32
    float*    Y    = (float*)carve(elems * 4);     // ssm output
    _Float16* xs16 = (_Float16*)carve(elems * 2);  // spike activations f16
    float*    dt   = (float*)carve((size_t)ROWS * R_ * 4);
    float*    dAb  = (float*)carve((size_t)ROWS * N_ * 4);
    float*    u    = (float*)carve((size_t)ROWS * N_ * 4);
    float*    dtm  = (float*)carve((size_t)ROWS * 4);
    _Float16* H16  = (_Float16*)carve((size_t)ROWS * N_ * 2);
    _Float16* Wdt16 = (_Float16*)carve((size_t)R_ * D_ * 2);
    _Float16* Bm16  = (_Float16*)carve((size_t)N_ * D_ * 2);
    _Float16* Cm16  = (_Float16*)carve((size_t)D_ * N_ * 2);

    // ---- zero persistent LIF states (v_i..a_o contiguous: 4*elems floats) ----
    zero_kernel<<<(4 * elems / 4) / 256, 256, 0, stream>>>((float4*)v_i);

    // ---- one-shot weight conversion to f16 ----
    f32_to_f16_kernel<<<(R_ * D_) / 256, 256, 0, stream>>>(Wdt, Wdt16);
    f32_to_f16_kernel<<<(N_ * D_) / 256, 256, 0, stream>>>(Bm,  Bm16);
    f32_to_f16_kernel<<<(D_ * N_) / 256, 256, 0, stream>>>(Cm,  Cm16);

    const int lif_blocks = (int)(elems / 256);
    for (int t = 0; t < T_; ++t) {
        // 1) input LIF -> xs (f32 + f16)
        lif_in_kernel<<<lif_blocks, 256, 0, stream>>>(
            x, t, v_i, a_i, tau_i, thr_i, rst_i, xs, xs16);

        // 2) dt = softplus(xs @ Wdt^T + bias)   [M=4096,N=64,K=1024]
        wmma_gemm_nt<0><<<dim3(ROWS / 128, 1), 256, 0, stream>>>(
            xs16, Wdt16, dt, dtb, nullptr, ROWS, R_, D_);

        // 3) dA = mean_r exp(dt*A), dtm = mean_r dt
        da_kernel<<<ROWS / 4, 256, 0, stream>>>(dt, A, dAb, dtm);

        // 4) u = xs @ Bm^T                      [M=4096,N=64,K=1024]
        wmma_gemm_nt<1><<<dim3(ROWS / 128, 1), 256, 0, stream>>>(
            xs16, Bm16, u, nullptr, nullptr, ROWS, N_, D_);

        // 5) sequential state scan over L (tiny)
        hscan_kernel<<<B_, 64, 0, stream>>>(dAb, dtm, u, H16);

        // 6) Y = H @ Cm^T + Dv*xs               [M=4096,N=1024,K=64]
        wmma_gemm_nt<2><<<dim3(ROWS / 128, D_ / 64), 256, 0, stream>>>(
            H16, Cm16, Y, Dv, xs, ROWS, D_, N_);

        // 7) output LIF -> d_out[:, t]
        lif_out_kernel<<<lif_blocks, 256, 0, stream>>>(
            Y, t, v_o, a_o, tau_o, thr_o, rst_o, out);
    }
}
